// NVFP4PackedLinear_30416958390322
// MI455X (gfx1250) — matrix operand order repaired and checked by reading, not compiled, not run on hardware
//
#include <hip/hip_runtime.h>
#include <math.h>

// ---------------------------------------------------------------------------
// NVFP4 packed linear + gelu(tanh) for gfx1250 (MI455X)
//   out[m,n] = gelu( sum_k x[m,k] * decode(W4[n,k]) * scale[n,k/32] + bias[n] )
// M=2048, K=4096, N=16384 (derived from in_sizes at launch).
// Strategy: f16 WMMA (v_wmma_f32_16x16x32_f16), per-32-block scale folded into
// the dequantized f16 weight during the LDS fill, f32 accumulation in WMMA C/D.
// ---------------------------------------------------------------------------

typedef __attribute__((ext_vector_type(16))) _Float16 v16h;
typedef __attribute__((ext_vector_type(8)))  _Float16 v8h;
typedef __attribute__((ext_vector_type(8)))  float    v8f;

#define TILE_M 128
#define TILE_N 128
#define KC      64   // K-chunk staged in LDS (= 2 scale blocks = 2 WMMA K-steps)

// e2m1 nibble -> IEEE f16 bit pattern (exact: 0,.5,1,1.5,2,3,4,6 with sign)
__device__ __forceinline__ unsigned fp4_to_f16bits(unsigned c) {
  unsigned s = (c >> 3) & 1u;
  unsigned e = (c >> 1) & 3u;
  unsigned m = c & 1u;
  unsigned mag = e ? (((14u + e) << 10) | (m << 9)) : (m ? 0x3800u : 0u);
  return (s << 15) | mag;
}

// one packed byte (2 nibbles) -> 2 scaled f16, packed into a dword
__device__ __forceinline__ unsigned dec_pair(unsigned byte, _Float16 hs) {
  unsigned short bl = (unsigned short)fp4_to_f16bits(byte & 15u);
  unsigned short bh = (unsigned short)fp4_to_f16bits((byte >> 4) & 15u);
  _Float16 hl = __builtin_bit_cast(_Float16, bl) * hs;   // low nibble -> even k
  _Float16 hh = __builtin_bit_cast(_Float16, bh) * hs;   // high nibble -> odd k
  union { _Float16 h[2]; unsigned u; } r;
  r.h[0] = hl; r.h[1] = hh;
  return r.u;
}

__device__ __forceinline__ float gelu_tanh(float v) {
  float u = 0.7978845608028654f * (v + 0.044715f * v * v * v);
  return 0.5f * v * (1.0f + tanhf(u));
}

__global__ __launch_bounds__(256)
void nvfp4_linear_gelu_wmma(const float* __restrict__ x,
                            const int*   __restrict__ wp,
                            const float* __restrict__ wscale,
                            const float* __restrict__ bias,
                            float*       __restrict__ out,
                            int M, int N, int K) {
  __shared__ _Float16 As[TILE_M][KC];   // x tile, f16        (16 KB)
  __shared__ _Float16 Bs[TILE_N][KC];   // scaled weights f16 (16 KB)

  const int tid  = threadIdx.x;        // 0..255, 8 wave32
  const int wid  = tid >> 5;
  const int lane = tid & 31;
  const int n0   = blockIdx.x * TILE_N;
  const int m0   = blockIdx.y * TILE_M;
  const int Kw   = K >> 1;             // packed words per weight row
  const int Kb   = K >> 5;             // scale blocks per weight row

  // 8 waves -> 2 (M) x 4 (N); each wave owns a 64x32 sub-tile = 4x2 WMMA tiles
  const int mwave = (wid >> 2) * 64;
  const int nwave = (wid &  3) * 32;
  const int lhalf = lane >> 4;         // 0: lanes 0-15, 1: lanes 16-31
  const int l15   = lane & 15;

  v8f acc[4][2];
  #pragma unroll
  for (int mt = 0; mt < 4; ++mt)
    #pragma unroll
    for (int nt = 0; nt < 2; ++nt)
      acc[mt][nt] = {};

  for (int kc = 0; kc < K; kc += KC) {
    __syncthreads();   // protect LDS reuse from previous iteration's readers

    // ---- stage A: 128x64 fp32 -> f16 (coalesced float4 loads) -------------
    #pragma unroll
    for (int i = 0; i < 8; ++i) {
      int idx = tid + 256 * i;         // 0..2047
      int m   = idx >> 4;              // 0..127
      int f4  = idx & 15;              // 16 float4 per 64-wide row
      const float* px = x + (size_t)(m0 + m) * K + kc + f4 * 4;
      float4 xa = *(const float4*)px;
      if (kc + KC < K) __builtin_prefetch(px + KC, 0, 0);   // global_prefetch_b8
      union { _Float16 h[4]; uint2 u; } pa;
      pa.h[0] = (_Float16)xa.x; pa.h[1] = (_Float16)xa.y;
      pa.h[2] = (_Float16)xa.z; pa.h[3] = (_Float16)xa.w;
      *(uint2*)&As[m][f4 * 4] = pa.u;
    }

    // ---- stage B: 128x(32 words) fp4 -> f16 * block scale -----------------
    #pragma unroll
    for (int i = 0; i < 4; ++i) {
      int idx = tid + 256 * i;         // 0..1023
      int n   = idx >> 3;              // 0..127
      int q   = idx & 7;               // 8 int4 per 32-word row
      const int* pw = wp + (size_t)(n0 + n) * Kw + (kc >> 1) + q * 4;
      int4 w4 = *(const int4*)pw;      // 4 packed bytes = 8 k-values
      if (kc + KC < K) __builtin_prefetch(pw + (KC >> 1), 0, 0);
      // this int4 covers k in [kc+q*8, kc+q*8+8) -> single 32-block scale
      float s = wscale[(size_t)(n0 + n) * Kb + (kc >> 5) + (q >> 2)];
      _Float16 hs = (_Float16)s;
      uint4 pk;
      pk.x = dec_pair((unsigned)w4.x & 0xFFu, hs);
      pk.y = dec_pair((unsigned)w4.y & 0xFFu, hs);
      pk.z = dec_pair((unsigned)w4.z & 0xFFu, hs);
      pk.w = dec_pair((unsigned)w4.w & 0xFFu, hs);
      *(uint4*)&Bs[n][q * 8] = pk;
    }

    __syncthreads();

    // ---- compute: 2 K-steps of 32, 4x2 WMMA tiles per wave ----------------
    #pragma unroll
    for (int kb = 0; kb < 2; ++kb) {
      v16h afrag[4], bfrag[2];

      // A 16x32 f16 layout: lanes 0-15 hold M=l, K {0..7,16..23};
      //                     lanes 16-31 hold M=l-16, K {8..15,24..31}
      #pragma unroll
      for (int mt = 0; mt < 4; ++mt) {
        int row = mwave + mt * 16 + l15;
        int ko  = kb * 32 + lhalf * 8;
        union { v16h v; v8h p[2]; } ua;
        ua.p[0] = *(const v8h*)&As[row][ko];        // ds_load_b128
        ua.p[1] = *(const v8h*)&As[row][ko + 16];   // ds_load_b128
        afrag[mt] = ua.v;
      }
      // B 32x16 f16 layout: lanes 0-15 hold col=l, K {0..7,16..23};
      //                     lanes 16-31 hold col=l-16, K {8..15,24..31}
      #pragma unroll
      for (int nt = 0; nt < 2; ++nt) {
        int col = nwave + nt * 16 + l15;
        int ko  = kb * 32 + lhalf * 8;
        union { v16h v; v8h p[2]; } ub;
        ub.p[0] = *(const v8h*)&Bs[col][ko];
        ub.p[1] = *(const v8h*)&Bs[col][ko + 16];
        bfrag[nt] = ub.v;
      }

      #pragma unroll
      for (int mt = 0; mt < 4; ++mt)
        #pragma unroll
        for (int nt = 0; nt < 2; ++nt)
          acc[mt][nt] = __builtin_amdgcn_wmma_f32_16x16x32_f16(
              false, afrag[mt], false, bfrag[nt],
              (short)0, acc[mt][nt], false, false);
    }
  }

  // ---- epilogue: bias + gelu; D layout: lane<16 -> rows 0-7, lane>=16 -> 8-15
  #pragma unroll
  for (int nt = 0; nt < 2; ++nt) {
    int gcol = n0 + nwave + nt * 16 + l15;
    float bv = bias[gcol];
    #pragma unroll
    for (int mt = 0; mt < 4; ++mt) {
      int grow = m0 + mwave + mt * 16 + lhalf * 8;
      #pragma unroll
      for (int j = 0; j < 8; ++j) {
        float v = acc[mt][nt][j] + bv;
        out[(size_t)(grow + j) * N + gcol] = gelu_tanh(v);
      }
    }
  }
}

extern "C" void kernel_launch(void* const* d_in, const int* in_sizes, int n_in,
                              void* d_out, int out_size, void* d_ws, size_t ws_size,
                              hipStream_t stream) {
  const float* x    = (const float*)d_in[0];   // [4,512,K] fp32
  const int*   wp   = (const int*)d_in[1];     // [N, K/2] packed bytes in int32
  const float* ws   = (const float*)d_in[2];   // [N, K/32] fp32
  const float* bias = (const float*)d_in[3];   // [N] fp32
  float* out = (float*)d_out;                  // [4,512,N] fp32

  const int N  = in_sizes[3];
  const int K  = (in_sizes[1] / N) * 2;
  const int M  = in_sizes[0] / K;

  dim3 grid(N / TILE_N, M / TILE_M);
  nvfp4_linear_gelu_wmma<<<grid, 256, 0, stream>>>(x, wp, ws, bias, out, M, N, K);
}
